// SpatialNet_22393959481842
// MI455X (gfx1250) — matrix operand order repaired and checked by reading, not compiled, not run on hardware
//
#include <hip/hip_runtime.h>
#include <hip/hip_bf16.h>
#include <math.h>

#define BS 64
#define STEP 128
#define NJ 25
#define DIM 256
#define CH 128              // DIM/2: channels of x
#define NCOL 3200           // NJ*STEP per batch
#define EPS 1e-5f

typedef __attribute__((ext_vector_type(16))) _Float16 v16h;
typedef __attribute__((ext_vector_type(8)))  _Float16 v8h;
typedef __attribute__((ext_vector_type(8)))  float    v8f;

// ---------------------------------------------------------------------------
// K0: zero the output (final GEMM max-reduces via bit-pattern atomicMax of
// non-negative floats; 0-init each launch keeps replays idempotent).
// ---------------------------------------------------------------------------
__global__ void init_out_kernel(float* out, int n) {
    int i = blockIdx.x * blockDim.x + threadIdx.x;
    if (i < n) out[i] = 0.0f;
}

// ---------------------------------------------------------------------------
// K1: spatial one-hot embedding: spa1[o][j] = relu(Ws2 @ relu(Ws1[:,j]+bs1) + bs2)
// ---------------------------------------------------------------------------
__global__ void spa_kernel(const float* __restrict__ Ws1, const float* __restrict__ bs1,
                           const float* __restrict__ Ws2, const float* __restrict__ bs2,
                           float* __restrict__ spa1) {
    __shared__ float h[64];
    int j = blockIdx.x;     // joint
    int o = threadIdx.x;    // channel
    float v = Ws1[o * NJ + j] + bs1[o];           // Ws1 @ e_j picks column j
    h[o] = v > 0.0f ? v : 0.0f;
    __syncthreads();
    float acc = bs2[o];
    for (int c = 0; c < 64; ++c) acc += Ws2[o * 64 + c] * h[c];
    spa1[o * NJ + j] = acc > 0.0f ? acc : 0.0f;   // layout [64][25]
}

// ---------------------------------------------------------------------------
// K2: embeds (BN -> 3->64 -> relu -> 64->64 -> relu) for input & dif, summed,
// plus spa broadcast; writes x (f16) into rows 128..255 of z (per-batch 256x3200).
// ---------------------------------------------------------------------------
__global__ void embed_kernel(const float* __restrict__ input, const float* __restrict__ dif,
                             const float* __restrict__ jg, const float* __restrict__ jb,
                             const float* __restrict__ jm, const float* __restrict__ jv,
                             const float* __restrict__ dg, const float* __restrict__ db,
                             const float* __restrict__ dm, const float* __restrict__ dv,
                             const float* __restrict__ Wj1, const float* __restrict__ bj1,
                             const float* __restrict__ Wj2, const float* __restrict__ bj2,
                             const float* __restrict__ Wd1, const float* __restrict__ bd1,
                             const float* __restrict__ Wd2, const float* __restrict__ bd2,
                             const float* __restrict__ spa1, _Float16* __restrict__ z) {
    int p = blockIdx.x * blockDim.x + threadIdx.x;
    int s = p % STEP;
    int t = p / STEP;
    int j = t % NJ;
    int b = t / NJ;

    float xin[3], xdn[3];
    #pragma unroll
    for (int c = 0; c < 3; ++c) {
        int ci  = c * NJ + j;                        // BatchNorm1d channel index
        int idx = ((b * 3 + c) * NJ + j) * STEP + s;
        float xi = input[idx];
        xin[c] = (xi - jm[ci]) * (jg[ci] * rsqrtf(jv[ci] + EPS)) + jb[ci];
        float xd = dif[idx];
        xdn[c] = (xd - dm[ci]) * (dg[ci] * rsqrtf(dv[ci] + EPS)) + db[ci];
    }

    float hp[64], hd[64];
    for (int o = 0; o < 64; ++o) {
        float a = bj1[o] + Wj1[o*3]*xin[0] + Wj1[o*3+1]*xin[1] + Wj1[o*3+2]*xin[2];
        hp[o] = fmaxf(a, 0.0f);
        float d2 = bd1[o] + Wd1[o*3]*xdn[0] + Wd1[o*3+1]*xdn[1] + Wd1[o*3+2]*xdn[2];
        hd[o] = fmaxf(d2, 0.0f);
    }

    _Float16* zb = z + (size_t)b * DIM * NCOL;
    int n = j * STEP + s;
    for (int o = 0; o < 64; ++o) {
        float ap = bj2[o], ad = bd2[o];
        for (int c = 0; c < 64; ++c) {
            ap += Wj2[o * 64 + c] * hp[c];
            ad += Wd2[o * 64 + c] * hd[c];
        }
        float e = fmaxf(ap, 0.0f) + fmaxf(ad, 0.0f);
        zb[(size_t)(CH + o) * NCOL + n]      = (_Float16)e;             // pos+d channels
        zb[(size_t)(CH + 64 + o) * NCOL + n] = (_Float16)spa1[o*NJ+j];  // spa channels
    }
}

// ---------------------------------------------------------------------------
// K3: convert/stack weights to f16 GEMM A-matrices:
//   A1 (512x128) = [Wg1; Wg2]     A2 (256x256) = [Ww | Ww1]
// ---------------------------------------------------------------------------
__global__ void wconv_kernel(const float* __restrict__ Wg1, const float* __restrict__ Wg2,
                             const float* __restrict__ Ww,  const float* __restrict__ Ww1,
                             _Float16* __restrict__ A1, _Float16* __restrict__ A2) {
    int i = blockIdx.x * blockDim.x + threadIdx.x;
    if (i < 512 * 128) {
        int m = i / 128, k = i % 128;
        float v = (m < 256) ? Wg1[m * 128 + k] : Wg2[(m - 256) * 128 + k];
        A1[i] = (_Float16)v;
    }
    if (i < 256 * 256) {
        int m = i / 256, k = i % 256;
        float v = (k < 128) ? Ww[m * 128 + k] : Ww1[m * 128 + (k - 128)];
        A2[i] = (_Float16)v;
    }
}

// ---------------------------------------------------------------------------
// WMMA GEMM core helpers.
//
// B staging: the 32(K) x 64(N) f16 tile is stored in LDS in *fragment order*:
//   element (k, n) of N-subtile nt = n/16 lives at
//     F[nt][ lane = (n%16) + 16*(k/16) ][ i = k%16 ]
// so each lane's 16-half B fragment is one contiguous, 32B-aligned v16h
// (lowered to ds_load_b128 x2) instead of 16 strided ds_load_u16.
//
// A fragment: lane (m = lane%16, half = lane/16)
// holds K in {half*16..half*16+15} (matching B fragment order): two aligned v8h
// global loads, shuffled into one v16h.
// ---------------------------------------------------------------------------
__device__ __forceinline__ void stage_b_tile(const _Float16* __restrict__ Bg,
                                             _Float16 (&F)[4][32][16],
                                             int k0, int n0, int K) {
    int t  = threadIdx.x;
    int kk = t >> 3;              // 0..31 : K row within tile
    int c0 = (t & 7) * 8;         // 0..56 : first of 8 contiguous N cols
    const _Float16* src = Bg + (size_t)(k0 + kk) * NCOL + n0 + c0;
    v8h vals = *(const v8h*)src;
    if (k0 + 32 < K) __builtin_prefetch(src + 32 * NCOL, 0, 0);  // global_prefetch_b8
    int klo = kk & 15;
    int khi = (kk >> 4) * 16;
    #pragma unroll
    for (int e = 0; e < 8; ++e) {
        int nc = c0 + e;
        F[nc >> 4][(nc & 15) + khi][klo] = vals[e];
    }
}

__device__ __forceinline__ v16h load_a_frag(const _Float16* __restrict__ Arow,
                                            int k0, int half) {
    v8h alo = *(const v8h*)(Arow + k0 + half * 16);
    v8h ahi = *(const v8h*)(Arow + k0 + half * 16 + 8);
    return __builtin_shufflevector(alo, ahi, 0, 1, 2, 3, 4, 5, 6, 7,
                                   8, 9, 10, 11, 12, 13, 14, 15);
}

// ---------------------------------------------------------------------------
// K4: GEMM1 (WMMA): qk[b] (512x3200 f16) = A1 (512x128) @ x_b (128x3200) + bias
// grid (3200/64, 512/128, 64); 256 threads = 8 waves; wave = 16(M) x 64(N) strip.
// ---------------------------------------------------------------------------
__global__ void __launch_bounds__(256) gemm_qk_kernel(const _Float16* __restrict__ A1,
                                                      const _Float16* __restrict__ z,
                                                      const float* __restrict__ bg1,
                                                      const float* __restrict__ bg2,
                                                      _Float16* __restrict__ qk) {
    __shared__ _Float16 F[4][32][16];
    int b    = blockIdx.z;
    int n0   = blockIdx.x * 64;
    int wave = threadIdx.x >> 5;
    int lane = threadIdx.x & 31;
    int m0   = blockIdx.y * 128 + wave * 16;
    int half = lane >> 4;
    int ml   = lane & 15;

    const _Float16* Bg   = z + (size_t)b * DIM * NCOL + (size_t)CH * NCOL; // x rows
    const _Float16* Arow = A1 + (size_t)(m0 + ml) * 128;

    v8f acc[4] = {};
    for (int k0 = 0; k0 < 128; k0 += 32) {
        stage_b_tile(Bg, F, k0, n0, 128);
        __syncthreads();
        v16h af = load_a_frag(Arow, k0, half);
        #pragma unroll
        for (int nt = 0; nt < 4; ++nt) {
            v16h bf = *(const v16h*)&F[nt][lane][0];
            acc[nt] = __builtin_amdgcn_wmma_f32_16x16x32_f16(false, af, false, bf,
                                                             (short)0, acc[nt],
                                                             false, false);
        }
        __syncthreads();
    }

    // epilogue: add bias (bg1 for q rows, bg2 for k rows), store f16
    _Float16* out = qk + (size_t)b * 512 * NCOL;
    #pragma unroll
    for (int nt = 0; nt < 4; ++nt) {
        int nc = n0 + nt * 16 + ml;
        #pragma unroll
        for (int v = 0; v < 8; ++v) {
            int m = m0 + v + 8 * half;       // C/D layout: M = v + 8*(lane/16)
            float bias = (m < 256) ? bg1[m] : bg2[m - 256];
            out[(size_t)m * NCOL + nc] = (_Float16)(acc[nt][v] + bias);
        }
    }
}

// ---------------------------------------------------------------------------
// K5: per-frame attention: scores = q_f^T k_f (25x25, K=256), softmax over k,
// agg[c][j] = sum_k g[j][k] x[c][k]; writes agg (f16) into rows 0..127 of z.
// ---------------------------------------------------------------------------
__global__ void __launch_bounds__(256) attn_kernel(const _Float16* __restrict__ qk,
                                                   _Float16* __restrict__ z) {
    __shared__ _Float16 qs[256][NJ];
    __shared__ _Float16 ks[256][NJ];
    __shared__ _Float16 xs[CH][NJ];
    __shared__ float g[NJ][NJ];
    int s = blockIdx.x, b = blockIdx.y;
    const _Float16* qkb = qk + (size_t)b * 512 * NCOL;
    _Float16*       zb  = z  + (size_t)b * DIM * NCOL;

    for (int i = threadIdx.x; i < 512 * NJ; i += 256) {
        int c = i / NJ, j = i % NJ;
        _Float16 vv = qkb[(size_t)c * NCOL + j * STEP + s];
        if (c < 256) qs[c][j] = vv; else ks[c - 256][j] = vv;
    }
    for (int i = threadIdx.x; i < CH * NJ; i += 256) {
        int c = i / NJ, j = i % NJ;
        xs[c][j] = zb[(size_t)(CH + c) * NCOL + j * STEP + s];
    }
    __syncthreads();

    for (int i = threadIdx.x; i < NJ * NJ; i += 256) {
        int j = i / NJ, kk = i % NJ;
        float acc = 0.0f;
        for (int c = 0; c < 256; ++c) acc += (float)qs[c][j] * (float)ks[c][kk];
        g[j][kk] = acc;
    }
    __syncthreads();

    if (threadIdx.x < NJ) {
        int j = threadIdx.x;
        float mx = g[j][0];
        for (int kk = 1; kk < NJ; ++kk) mx = fmaxf(mx, g[j][kk]);
        float e[NJ], sum = 0.0f;
        for (int kk = 0; kk < NJ; ++kk) { e[kk] = __expf(g[j][kk] - mx); sum += e[kk]; }
        float inv = 1.0f / sum;
        for (int kk = 0; kk < NJ; ++kk) g[j][kk] = e[kk] * inv;
    }
    __syncthreads();

    for (int i = threadIdx.x; i < CH * NJ; i += 256) {
        int c = i / NJ, j = i % NJ;
        float acc = 0.0f;
        for (int kk = 0; kk < NJ; ++kk) acc += g[j][kk] * (float)xs[c][kk];
        zb[(size_t)c * NCOL + j * STEP + s] = (_Float16)acc;   // agg rows 0..127
    }
}

// ---------------------------------------------------------------------------
// K6: GEMM2 (WMMA): y = A2 (256x256) @ [agg; x]_b (256x3200) + bw1, then BN,
// ReLU, and fused max-over-joints via bit-pattern atomicMax (y >= 0 after ReLU).
// grid (3200/64, 256/128, 64); 256 threads = 8 waves; wave = 16(M) x 64(N) strip.
// ---------------------------------------------------------------------------
__global__ void __launch_bounds__(256) gemm_y_kernel(const _Float16* __restrict__ A2,
                                                     const _Float16* __restrict__ z,
                                                     const float* __restrict__ bw1,
                                                     const float* __restrict__ gg,
                                                     const float* __restrict__ gb,
                                                     const float* __restrict__ gm,
                                                     const float* __restrict__ gv,
                                                     float* __restrict__ out) {
    __shared__ _Float16 F[4][32][16];
    int b    = blockIdx.z;
    int n0   = blockIdx.x * 64;
    int wave = threadIdx.x >> 5;
    int lane = threadIdx.x & 31;
    int m0   = blockIdx.y * 128 + wave * 16;
    int half = lane >> 4;
    int ml   = lane & 15;

    const _Float16* Bg   = z + (size_t)b * DIM * NCOL;
    const _Float16* Arow = A2 + (size_t)(m0 + ml) * 256;

    v8f acc[4] = {};
    for (int k0 = 0; k0 < 256; k0 += 32) {
        stage_b_tile(Bg, F, k0, n0, 256);
        __syncthreads();
        v16h af = load_a_frag(Arow, k0, half);
        #pragma unroll
        for (int nt = 0; nt < 4; ++nt) {
            v16h bf = *(const v16h*)&F[nt][lane][0];
            acc[nt] = __builtin_amdgcn_wmma_f32_16x16x32_f16(false, af, false, bf,
                                                             (short)0, acc[nt],
                                                             false, false);
        }
        __syncthreads();
    }

    unsigned* outu = (unsigned*)out;
    #pragma unroll
    for (int nt = 0; nt < 4; ++nt) {
        int nc = n0 + nt * 16 + ml;
        int s  = nc & (STEP - 1);          // n = j*STEP + s
        #pragma unroll
        for (int v = 0; v < 8; ++v) {
            int m = m0 + v + 8 * half;
            float y = acc[nt][v] + bw1[m];
            y = (y - gm[m]) * (gg[m] * rsqrtf(gv[m] + EPS)) + gb[m];
            y = fmaxf(y, 0.0f);
            // max over joints: non-negative floats order identically as uint bits
            atomicMax(&outu[((size_t)b * DIM + m) * STEP + s], __float_as_uint(y));
        }
    }
}

// ---------------------------------------------------------------------------
extern "C" void kernel_launch(void* const* d_in, const int* in_sizes, int n_in,
                              void* d_out, int out_size, void* d_ws, size_t ws_size,
                              hipStream_t stream) {
    const float* input = (const float*)d_in[0];
    const float* dif   = (const float*)d_in[1];
    const float* jn_g  = (const float*)d_in[2];
    const float* jn_b  = (const float*)d_in[3];
    const float* jn_m  = (const float*)d_in[4];
    const float* jn_v  = (const float*)d_in[5];
    const float* dn_g  = (const float*)d_in[6];
    const float* dn_b  = (const float*)d_in[7];
    const float* dn_m  = (const float*)d_in[8];
    const float* dn_v  = (const float*)d_in[9];
    const float* Wj1   = (const float*)d_in[10];
    const float* bj1   = (const float*)d_in[11];
    const float* Wj2   = (const float*)d_in[12];
    const float* bj2   = (const float*)d_in[13];
    const float* Wd1   = (const float*)d_in[14];
    const float* bd1   = (const float*)d_in[15];
    const float* Wd2   = (const float*)d_in[16];
    const float* bd2   = (const float*)d_in[17];
    const float* Ws1   = (const float*)d_in[18];
    const float* bs1   = (const float*)d_in[19];
    const float* Ws2   = (const float*)d_in[20];
    const float* bs2   = (const float*)d_in[21];
    const float* Wg1   = (const float*)d_in[22];
    const float* bg1   = (const float*)d_in[23];
    const float* Wg2   = (const float*)d_in[24];
    const float* bg2   = (const float*)d_in[25];
    const float* Ww    = (const float*)d_in[26];
    const float* Ww1   = (const float*)d_in[27];
    const float* bw1   = (const float*)d_in[28];
    const float* gbn_g = (const float*)d_in[29];
    const float* gbn_b = (const float*)d_in[30];
    const float* gbn_m = (const float*)d_in[31];
    const float* gbn_v = (const float*)d_in[32];
    float* out = (float*)d_out;

    // workspace carve-up
    char* ws = (char*)d_ws;
    size_t off = 0;
    _Float16* z   = (_Float16*)(ws + off); off += (size_t)BS * DIM * NCOL * 2;   // 104.9 MB
    _Float16* qk  = (_Float16*)(ws + off); off += (size_t)BS * 512 * NCOL * 2;   // 209.7 MB
    _Float16* A1  = (_Float16*)(ws + off); off += (size_t)512 * 128 * 2;
    _Float16* A2  = (_Float16*)(ws + off); off += (size_t)256 * 256 * 2;
    float*    spa1= (float*)   (ws + off); off += (size_t)64 * NJ * 4;

    // K0: zero output (atomicMax accumulator)
    int nout = BS * DIM * STEP;
    init_out_kernel<<<(nout + 255) / 256, 256, 0, stream>>>(out, nout);

    // K1: spatial embedding table
    spa_kernel<<<NJ, 64, 0, stream>>>(Ws1, bs1, Ws2, bs2, spa1);

    // K3: weight conversion (independent of K1/K2)
    wconv_kernel<<<(512 * 128 + 255) / 256, 256, 0, stream>>>(Wg1, Wg2, Ww, Ww1, A1, A2);

    // K2: embeds -> x (f16) into z rows 128..255
    embed_kernel<<<(BS * NJ * STEP) / 256, 256, 0, stream>>>(
        input, dif, jn_g, jn_b, jn_m, jn_v, dn_g, dn_b, dn_m, dn_v,
        Wj1, bj1, Wj2, bj2, Wd1, bd1, Wd2, bd2, spa1, z);

    // K4: q,k = [Wg1;Wg2] @ x  (WMMA)
    gemm_qk_kernel<<<dim3(NCOL / 64, 512 / 128, BS), 256, 0, stream>>>(A1, z, bg1, bg2, qk);

    // K5: per-frame attention -> agg (f16) into z rows 0..127
    attn_kernel<<<dim3(STEP, BS), 256, 0, stream>>>(qk, z);

    // K6: y = [Ww|Ww1] @ [agg;x] + bw1, BN, ReLU, max over joints (WMMA + atomicMax)
    gemm_y_kernel<<<dim3(NCOL / 64, 256 / 128, BS), 256, 0, stream>>>(
        A2, z, bw1, gbn_g, gbn_b, gbn_m, gbn_v, out);
}
